// MinimalGPT_8778913153447
// MI455X (gfx1250) — compile-verified, matrix operand order checked by reading, not run on hardware
//
#include <hip/hip_runtime.h>

// ---------------- model constants ----------------
constexpr int V_ = 32000;
constexpr int D_ = 768;
constexpr int H_ = 12;
constexpr int L_ = 6;
constexpr int F_ = 3072;
constexpr int S_ = 1024;
constexpr int B_ = 2;
constexpr int DK_ = D_ / H_;          // 64
constexpr int M_ = B_ * S_;           // 2048 tokens

typedef __attribute__((ext_vector_type(16))) __bf16 v16bf;
typedef __attribute__((ext_vector_type(8)))  __bf16 v8bf;
typedef __attribute__((ext_vector_type(8)))  float  v8f;
typedef __attribute__((ext_vector_type(4)))  unsigned int u32x4;
typedef __attribute__((ext_vector_type(8)))  int i32x8;
typedef __attribute__((ext_vector_type(4)))  int i32x4;

__device__ __forceinline__ v8f wmma_bf16(v16bf a, v16bf b, v8f c) {
  return __builtin_amdgcn_wmma_f32_16x16x32_bf16(
      /*neg_a=*/false, a, /*neg_b=*/false, b,
      /*c_mod=*/(short)0, c, /*reuse_a=*/false, /*reuse_b=*/false);
}

// Fragment loader for 16-bit WMMA operands stored fragment-major ([row][k],
// 32 k-elements contiguous per row): two b128 loads per lane.
__device__ __forceinline__ v16bf load_frag(const __bf16* p, int kb) {
  v8bf lo = *(const v8bf*)(p + kb);
  v8bf hi = *(const v8bf*)(p + kb + 16);
  return __builtin_shufflevector(lo, hi, 0, 1, 2, 3, 4, 5, 6, 7,
                                 8, 9, 10, 11, 12, 13, 14, 15);
}

__device__ __forceinline__ void store_out(float* Y, size_t idx, float v) {
  Y[idx] = v;
}
__device__ __forceinline__ void store_out(__bf16* Y, size_t idx, float v) {
  Y[idx] = (__bf16)v;
}

// TDM: DMA one W-tile (tile_dim1=64 columns x tile_dim0=32 contiguous
// k-elements of bf16) from the pre-transposed weight matrix WT[N][K] into LDS.
// Destination layout in LDS is exactly fragment-major [col][k].
// D# packing per CDNA5 ISA.
__device__ __forceinline__ void tdm_load_w_tile(const __bf16* gptr,
                                                unsigned lds_addr, int K, int N) {
  unsigned long long ga = (unsigned long long)(uintptr_t)gptr;
  u32x4 g0;
  g0[0] = 1u;                                    // count=1, user mode
  g0[1] = lds_addr;                              // LDS byte address
  g0[2] = (unsigned)ga;                          // global_addr[31:0]
  g0[3] = (unsigned)((ga >> 32) & 0x1FFFFFFu) | (2u << 30);  // addr[56:32] | type=2
  i32x8 g1;
  g1[0] = (int)(1u << 16);                       // data_size=1 -> 2 bytes
  g1[1] = (int)(((unsigned)K & 0xFFFFu) << 16);  // tensor_dim0[15:0] @ bit48
  g1[2] = (int)((((unsigned)K >> 16) & 0xFFFFu) |
                (((unsigned)N & 0xFFFFu) << 16));  // dim0 hi | dim1 lo
  g1[3] = (int)((((unsigned)N >> 16) & 0xFFFFu) | (32u << 16));  // dim1 hi | tile_dim0=32
  g1[4] = 64;                                    // tile_dim1=64, tile_dim2=0
  g1[5] = K;                                     // tensor_dim0_stride[31:0]
  g1[6] = 0;                                     // stride hi | dim1_stride lo
  g1[7] = 0;
  i32x4 z4 = {0, 0, 0, 0};
#if defined(__clang_major__) && (__clang_major__ >= 23)
  i32x8 z8 = {0, 0, 0, 0, 0, 0, 0, 0};
  __builtin_amdgcn_tensor_load_to_lds(g0, g1, z4, z4, z8, 0);
#else
  __builtin_amdgcn_tensor_load_to_lds(g0, g1, z4, z4, 0);
#endif
}

// ---------------- weight fp32 -> bf16 with transpose ----------------
// in: [K][N] fp32 ; out: [N][K] bf16. Tiled for coalescing on both sides.
__global__ __launch_bounds__(256) void tcvt_kernel(const float* __restrict__ in,
                                                   __bf16* __restrict__ out,
                                                   int K, int N) {
  __shared__ float tile[32][33];
  const int n0 = blockIdx.x * 32, k0 = blockIdx.y * 32;
  const int tx = threadIdx.x;       // 0..31
  const int ty = threadIdx.y;       // 0..7
#pragma unroll
  for (int r = 0; r < 32; r += 8)
    tile[ty + r][tx] = in[(size_t)(k0 + ty + r) * N + n0 + tx];
  __syncthreads();
#pragma unroll
  for (int r = 0; r < 32; r += 8)
    out[(size_t)(n0 + ty + r) * K + k0 + tx] = (__bf16)tile[tx][ty + r];
}

// ---------------- embedding + positional ----------------
__global__ __launch_bounds__(256) void embed_kernel(
    const int* __restrict__ tokens, const float* __restrict__ emb,
    const float* __restrict__ pe, float* __restrict__ x) {
  int t = blockIdx.x;                 // 0..M_-1
  int tok = tokens[t];
  int s = t % S_;
  const float sc = 27.712812921102035f;  // sqrt(768)
#pragma unroll
  for (int q = 0; q < 3; ++q) {
    int c = threadIdx.x + q * 256;
    x[(size_t)t * D_ + c] = emb[(size_t)tok * D_ + c] * sc + pe[(size_t)s * D_ + c];
  }
}

// ---------------- tiled bf16 WMMA GEMM: Y = act(X @ W + bias) ----------------
// X: [M,K] IT (converted to bf16 in LDS); WT: [N,K] bf16 (pre-transposed);
// Y: [M,N] OT. Block: 256 threads = 8 waves; tile 64x64; K chunked by 32.
// W tile fetched by the Tensor Data Mover straight into fragment-major LDS.
template <typename IT, typename OT>
__global__ __launch_bounds__(256) void gemm_kernel(
    const IT* __restrict__ X, const __bf16* __restrict__ WT,
    const float* __restrict__ bias, OT* __restrict__ Y,
    int M, int N, int K, int relu) {
  __shared__ __align__(16) __bf16 Xs[64 * 32];   // [row][k]
  __shared__ __align__(16) __bf16 Wt[64 * 32];   // [col][k] (TDM destination)

  const int tid  = threadIdx.x;
  const int lane = tid & 31;
  const int w    = tid >> 5;
  const int mr   = w & 3;             // row subtile 0..3
  const int nh   = w >> 2;            // col half 0..1
  const int n0   = blockIdx.x * 64;
  const int m0   = blockIdx.y * 64;
  const int l15  = lane & 15;
  const int hi   = lane >> 4;
  const int kb   = hi * 8;
  const unsigned wt_lds = (unsigned)(uintptr_t)(&Wt[0]);

  v8f c0 = {};
  v8f c1 = {};

  const int xr = tid >> 2, xc0 = (tid & 3) * 8;   // X: 4 thr/row, 8 cols each

  for (int kk = 0; kk < K; kk += 32) {
    // W tile: single TDM descriptor issued by wave 0 (EXEC-independent DMA)
    if (tid == 0)
      tdm_load_w_tile(WT + (size_t)n0 * K + kk, wt_lds, K, N);

    // X tile 64x32: contiguous 8-element read per thread, one v8bf LDS store
    {
      const IT* xp = X + (size_t)(m0 + xr) * K + kk + xc0;
      v8bf xv;
#pragma unroll
      for (int j = 0; j < 8; ++j) xv[j] = (__bf16)xp[j];
      *(v8bf*)(&Xs[xr * 32 + xc0]) = xv;
    }
    if (kk + 32 < K) {  // global_prefetch_b8 for next tiles
      __builtin_prefetch(&WT[(size_t)(n0 + (tid & 63)) * K + kk + 32], 0, 1);
      __builtin_prefetch(&X[(size_t)(m0 + xr) * K + kk + 32 + xc0], 0, 1);
    }
    if (tid == 0)
      __builtin_amdgcn_s_wait_tensorcnt(0);  // W tile resident in LDS
    __syncthreads();

    v16bf a  = load_frag(&Xs[(mr * 16 + l15) * 32], kb);
    v16bf b0 = load_frag(&Wt[(nh * 32 + l15) * 32], kb);
    v16bf b1 = load_frag(&Wt[(nh * 32 + 16 + l15) * 32], kb);
    c0 = wmma_bf16(a, b0, c0);
    c1 = wmma_bf16(a, b1, c1);
    __syncthreads();
  }

  const int colA = n0 + nh * 32 + l15;
  const int colB = colA + 16;
  const float biasA = bias[colA];
  const float biasB = bias[colB];
  const int rbase = m0 + mr * 16 + hi * 8;
#pragma unroll
  for (int i = 0; i < 8; ++i) {
    int row = rbase + i;
    float ya = c0[i] + biasA;
    float yb = c1[i] + biasB;
    if (relu) { ya = fmaxf(ya, 0.f); yb = fmaxf(yb, 0.f); }
    store_out(Y, (size_t)row * N + colA, ya);
    store_out(Y, (size_t)row * N + colB, yb);
  }
}

// ---------------- flash-style causal attention ----------------
// One wave per (batch, head, 16-query tile). Keys streamed 32 at a time.
// Q/K/V are bf16 (written directly by the QKV GEMMs).
__global__ __launch_bounds__(256) void attn_kernel(
    const __bf16* __restrict__ qb, const __bf16* __restrict__ kbuf,
    const __bf16* __restrict__ vb, float* __restrict__ ob) {
  __shared__ __align__(16) __bf16 Pst[8][16 * 32];  // per-wave P staging

  const int tid  = threadIdx.x;
  const int lane = tid & 31;
  const int w    = tid >> 5;
  const int gw   = blockIdx.x * 8 + w;
  const int QT   = S_ / 16;
  const int qt   = gw % QT;
  const int h    = (gw / QT) % H_;
  const int b    = gw / (QT * H_);
  const int q0   = qt * 16;
  const int l15  = lane & 15;
  const int hi   = lane >> 4;
  const int kb   = hi * 8;

  v16bf aq[2];
#pragma unroll
  for (int dc = 0; dc < 2; ++dc)
    aq[dc] = load_frag(qb + ((size_t)(b * S_ + q0 + l15)) * D_ + h * DK_ + dc * 32, kb);

  v8f co[4] = {};                      // O accumulators: 16x64
  float m[8], lsum[8];
#pragma unroll
  for (int i = 0; i < 8; ++i) { m[i] = -3.0e38f; lsum[i] = 0.f; }

  const int nch = (q0 + 16 + 31) / 32;
  for (int ch = 0; ch < nch; ++ch) {
    const int jj = ch * 32;
    v8f sacc[2] = {};
#pragma unroll
    for (int half = 0; half < 2; ++half) {
      const int j0 = jj + half * 16;
#pragma unroll
      for (int dc = 0; dc < 2; ++dc) {
        v16bf bk = load_frag(
            kbuf + ((size_t)(b * S_ + j0 + l15)) * D_ + h * DK_ + dc * 32, kb);
        sacc[half] = wmma_bf16(aq[dc], bk, sacc[half]);
      }
    }

    // scale + causal mask (C-layout: row = i + 8*hi, col = lane&15)
    float sv[2][8];
#pragma unroll
    for (int half = 0; half < 2; ++half)
#pragma unroll
      for (int i = 0; i < 8; ++i) {
        float s = sacc[half][i] * 0.125f;      // 1/sqrt(64)
        int j = jj + half * 16 + l15;
        int qrow = q0 + i + 8 * hi;
        sv[half][i] = (j > qrow) ? -1.0e30f : s;
      }

    // online softmax update per row (reduce across each 16-lane group)
    float sc_[8];
#pragma unroll
    for (int i = 0; i < 8; ++i) {
      float mx = fmaxf(sv[0][i], sv[1][i]);
#pragma unroll
      for (int msk = 1; msk < 16; msk <<= 1) mx = fmaxf(mx, __shfl_xor(mx, msk, 16));
      float nm = fmaxf(m[i], mx);
      float scal = __expf(m[i] - nm);
      float p0 = __expf(sv[0][i] - nm);
      float p1 = __expf(sv[1][i] - nm);
      float ps = p0 + p1;
#pragma unroll
      for (int msk = 1; msk < 16; msk <<= 1) ps += __shfl_xor(ps, msk, 16);
      lsum[i] = lsum[i] * scal + ps;
      m[i] = nm;
      sc_[i] = scal;
      int prow = i + 8 * hi;
      Pst[w][prow * 32 + l15]      = (__bf16)p0;
      Pst[w][prow * 32 + 16 + l15] = (__bf16)p1;
    }
#pragma unroll
    for (int n = 0; n < 4; ++n)
#pragma unroll
      for (int i = 0; i < 8; ++i) co[n][i] *= sc_[i];

    asm volatile("s_wait_dscnt 0" ::: "memory");  // P staged in LDS -> readable

    v16bf ap = load_frag(&Pst[w][l15 * 32], kb);
#pragma unroll
    for (int n = 0; n < 4; ++n) {
      // V fragment: element offsets are compile-time immediates (stride D_)
      const __bf16* vp =
          vb + ((size_t)(b * S_ + jj + kb)) * D_ + h * DK_ + n * 16 + l15;
      v16bf bvf;
#pragma unroll
      for (int e = 0; e < 8; ++e) {
        bvf[e]     = vp[(size_t)e * D_];
        bvf[e + 8] = vp[(size_t)(16 + e) * D_];
      }
      co[n] = wmma_bf16(ap, bvf, co[n]);
    }
  }

#pragma unroll
  for (int n = 0; n < 4; ++n)
#pragma unroll
    for (int i = 0; i < 8; ++i) {
      int row = q0 + i + 8 * hi;
      ob[((size_t)(b * S_ + row)) * D_ + h * DK_ + n * 16 + l15] = co[n][i] / lsum[i];
    }
}

// ---------------- fused residual + LayerNorm ----------------
__global__ __launch_bounds__(256) void ln_kernel(
    const float* __restrict__ x, const float* __restrict__ r,
    const float* __restrict__ g, const float* __restrict__ beta,
    float* __restrict__ y) {
  __shared__ float red[256];
  const int row = blockIdx.x;
  const int tid = threadIdx.x;
  const float* xp = x + (size_t)row * D_;
  const float* rp = r ? r + (size_t)row * D_ : nullptr;

  float v[3];
  float s = 0.f;
#pragma unroll
  for (int q = 0; q < 3; ++q) {
    int c = tid + q * 256;
    float t = xp[c] + (rp ? rp[c] : 0.f);
    v[q] = t;
    s += t;
  }
  red[tid] = s;
  __syncthreads();
  for (int st = 128; st > 0; st >>= 1) {
    if (tid < st) red[tid] += red[tid + st];
    __syncthreads();
  }
  const float mean = red[0] * (1.0f / D_);
  __syncthreads();

  float s2 = 0.f;
#pragma unroll
  for (int q = 0; q < 3; ++q) { float d = v[q] - mean; s2 += d * d; }
  red[tid] = s2;
  __syncthreads();
  for (int st = 128; st > 0; st >>= 1) {
    if (tid < st) red[tid] += red[tid + st];
    __syncthreads();
  }
  const float inv = rsqrtf(red[0] * (1.0f / D_) + 1e-5f);

#pragma unroll
  for (int q = 0; q < 3; ++q) {
    int c = tid + q * 256;
    y[(size_t)row * D_ + c] = (v[q] - mean) * inv * g[c] + beta[c];
  }
}

// ---------------- host orchestration ----------------
extern "C" void kernel_launch(void* const* d_in, const int* in_sizes, int n_in,
                              void* d_out, int out_size, void* d_ws, size_t ws_size,
                              hipStream_t stream) {
  (void)in_sizes; (void)n_in; (void)out_size; (void)ws_size;

  const int*   tokens = (const int*)  d_in[0];
  const float* emb    = (const float*)d_in[1];
  const float* pe     = (const float*)d_in[2];
  const float* wq     = (const float*)d_in[3];
  const float* bq     = (const float*)d_in[4];
  const float* wk     = (const float*)d_in[5];
  const float* bk     = (const float*)d_in[6];
  const float* wv     = (const float*)d_in[7];
  const float* bv     = (const float*)d_in[8];
  const float* wo     = (const float*)d_in[9];
  const float* bo     = (const float*)d_in[10];
  const float* w1     = (const float*)d_in[11];
  const float* b1     = (const float*)d_in[12];
  const float* w2     = (const float*)d_in[13];
  const float* b2     = (const float*)d_in[14];
  const float* g1     = (const float*)d_in[15];
  const float* be1    = (const float*)d_in[16];
  const float* g2     = (const float*)d_in[17];
  const float* be2    = (const float*)d_in[18];
  const float* gf     = (const float*)d_in[19];
  const float* bff    = (const float*)d_in[20];
  const float* wout   = (const float*)d_in[21];
  const float* bout   = (const float*)d_in[22];
  float* out = (float*)d_out;

  char* ws = (char*)d_ws;
  size_t off = 0;
  auto take = [&](size_t bytes) {
    size_t o = off;
    off += (bytes + 255) & ~(size_t)255;
    return o;
  };
  const size_t DD = (size_t)L_ * D_ * D_;
  const size_t DF = (size_t)L_ * D_ * F_;
  __bf16* wq_bf   = (__bf16*)(ws + take(DD * 2));   // [N][K] transposed bf16
  __bf16* wk_bf   = (__bf16*)(ws + take(DD * 2));
  __bf16* wv_bf   = (__bf16*)(ws + take(DD * 2));
  __bf16* wo_bf   = (__bf16*)(ws + take(DD * 2));
  __bf16* w1_bf   = (__bf16*)(ws + take(DF * 2));
  __bf16* w2_bf   = (__bf16*)(ws + take(DF * 2));
  __bf16* wout_bf = (__bf16*)(ws + take((size_t)D_ * V_ * 2));
  float*  x     = (float*) (ws + take(4ull * M_ * D_));
  __bf16* qact  = (__bf16*)(ws + take(2ull * M_ * D_));
  __bf16* kact  = (__bf16*)(ws + take(2ull * M_ * D_));
  __bf16* vact  = (__bf16*)(ws + take(2ull * M_ * D_));
  float*  attno = (float*) (ws + take(4ull * M_ * D_));
  float*  t1    = (float*) (ws + take(4ull * M_ * D_));
  __bf16* ff1   = (__bf16*)(ws + take(2ull * M_ * F_));

  // transpose-convert each per-layer weight slice: in [K][N] -> out [N][K]
  auto tcvt = [&](const float* in, __bf16* o, int K, int N) {
    tcvt_kernel<<<dim3(N / 32, K / 32), dim3(32, 8), 0, stream>>>(in, o, K, N);
  };
  for (int l = 0; l < L_; ++l) {
    tcvt(wq + (size_t)l * D_ * D_, wq_bf + (size_t)l * D_ * D_, D_, D_);
    tcvt(wk + (size_t)l * D_ * D_, wk_bf + (size_t)l * D_ * D_, D_, D_);
    tcvt(wv + (size_t)l * D_ * D_, wv_bf + (size_t)l * D_ * D_, D_, D_);
    tcvt(wo + (size_t)l * D_ * D_, wo_bf + (size_t)l * D_ * D_, D_, D_);
    tcvt(w1 + (size_t)l * D_ * F_, w1_bf + (size_t)l * D_ * F_, D_, F_);
    tcvt(w2 + (size_t)l * F_ * D_, w2_bf + (size_t)l * F_ * D_, F_, D_);
  }
  tcvt(wout, wout_bf, D_, V_);

  embed_kernel<<<M_, 256, 0, stream>>>(tokens, emb, pe, x);

  const dim3 gDD(D_ / 64, M_ / 64);
  const dim3 gDF(F_ / 64, M_ / 64);
  const dim3 gDV(V_ / 64, M_ / 64);
  const int attn_blocks = (B_ * H_ * (S_ / 16)) / 8;

  for (int l = 0; l < L_; ++l) {
    const __bf16* Wq = wq_bf + (size_t)l * D_ * D_;
    const __bf16* Wk = wk_bf + (size_t)l * D_ * D_;
    const __bf16* Wv = wv_bf + (size_t)l * D_ * D_;
    const __bf16* Wo = wo_bf + (size_t)l * D_ * D_;
    const __bf16* W1 = w1_bf + (size_t)l * D_ * F_;
    const __bf16* W2 = w2_bf + (size_t)l * F_ * D_;

    gemm_kernel<float, __bf16><<<gDD, 256, 0, stream>>>(
        x, Wq, bq + (size_t)l * D_, qact, M_, D_, D_, 0);
    gemm_kernel<float, __bf16><<<gDD, 256, 0, stream>>>(
        x, Wk, bk + (size_t)l * D_, kact, M_, D_, D_, 0);
    gemm_kernel<float, __bf16><<<gDD, 256, 0, stream>>>(
        x, Wv, bv + (size_t)l * D_, vact, M_, D_, D_, 0);

    attn_kernel<<<attn_blocks, 256, 0, stream>>>(qact, kact, vact, attno);

    gemm_kernel<float, float><<<gDD, 256, 0, stream>>>(
        attno, Wo, bo + (size_t)l * D_, t1, M_, D_, D_, 0);
    ln_kernel<<<M_, 256, 0, stream>>>(x, t1, g1 + (size_t)l * D_,
                                      be1 + (size_t)l * D_, x);

    gemm_kernel<float, __bf16><<<gDF, 256, 0, stream>>>(
        x, W1, b1 + (size_t)l * F_, ff1, M_, F_, D_, 1);
    gemm_kernel<__bf16, float><<<gDD, 256, 0, stream>>>(
        ff1, W2, b2 + (size_t)l * D_, t1, M_, D_, F_, 0);
    ln_kernel<<<M_, 256, 0, stream>>>(x, t1, g2 + (size_t)l * D_,
                                      be2 + (size_t)l * D_, x);
  }

  ln_kernel<<<M_, 256, 0, stream>>>(x, nullptr, gf, bff, t1);
  gemm_kernel<float, float><<<gDV, 256, 0, stream>>>(
      t1, wout_bf, bout, out, M_, V_, D_, 0);
}
